// SubstitutionHead_31379031064688
// MI455X (gfx1250) — matrix operand order repaired and checked by reading, not compiled, not run on hardware
//
#include <hip/hip_runtime.h>
#include <hip/hip_bf16.h>
#include <stdint.h>

typedef __attribute__((ext_vector_type(16))) _Float16 v16h;
typedef __attribute__((ext_vector_type(8)))  float    v8f;

#define Bn   8
#define Tcn  1024
#define En   256
#define Kk   8
#define Mm   2048
#define Vp   17                      // V+1
#define T1n  (Tcn*Kk)                // 8192
#define TTn  (T1n + Mm*Kk)           // 24576 tokens total per batch
#define NPAD 144                     // Kk*Vp=136 padded to 9 N-tiles
#define MAXTILES (Bn*(Mm/16 + Kk))   // worst case tile count = 1088
#define NT   512

// ---------------------------------------------------------------- init / prep
__global__ void k_init(uint32_t* __restrict__ x0h32, int* __restrict__ tileCnt) {
    size_t n = (size_t)Bn * Mm * En / 2;   // x0h as dwords
    for (size_t i = (size_t)blockIdx.x * blockDim.x + threadIdx.x; i < n;
         i += (size_t)gridDim.x * blockDim.x)
        x0h32[i] = 0u;
    if (blockIdx.x == 0 && threadIdx.x == 0) *tileCnt = 0;
}

__global__ __launch_bounds__(1024) void k_maxdepth(const int* __restrict__ depth,
                                                   int* __restrict__ maxd) {
    __shared__ int red[1024];
    int tid = threadIdx.x;
    int mv = (int)0x80000000;
    for (int i = tid; i < Bn * TTn; i += 1024) {
        int d = depth[i];
        mv = d > mv ? d : mv;
    }
    red[tid] = mv; __syncthreads();
    for (int s = 512; s > 0; s >>= 1) {
        if (tid < s) red[tid] = red[tid] > red[tid + s] ? red[tid] : red[tid + s];
        __syncthreads();
    }
    if (tid == 0) *maxd = red[0];
}

__global__ void k_cvt_x(const float* __restrict__ x, _Float16* __restrict__ xh) {
    size_t n = (size_t)Bn * Tcn * En;
    for (size_t i = (size_t)blockIdx.x * blockDim.x + threadIdx.x; i < n;
         i += (size_t)gridDim.x * blockDim.x)
        xh[i] = (_Float16)x[i];
}

// Whb[j][o][e] = W1[e][o][j]  (B-matrix transposed: K(=e)-contiguous per column o)
__global__ void k_cvt_w1(const float* __restrict__ W1, _Float16* __restrict__ Whb) {
    size_t n = (size_t)Kk * En * En;
    for (size_t i = (size_t)blockIdx.x * blockDim.x + threadIdx.x; i < n;
         i += (size_t)gridDim.x * blockDim.x) {
        int j = (int)(i / (En * En));
        int rem = (int)(i % (En * En));
        int o = rem / En, e = rem % En;
        Whb[i] = (_Float16)W1[((size_t)e * En + o) * Kk + j];
    }
}

// WcT[jv][e] = sum_o W0[e][o][j] * Wl[v][o],  jv = j*17+v  (padded to 144 rows)
__global__ void k_build_wc(const float* __restrict__ W0, const float* __restrict__ Wl,
                           _Float16* __restrict__ WcT) {
    int i = blockIdx.x * blockDim.x + threadIdx.x;
    if (i >= NPAD * En) return;
    int jv = i / En, e = i % En;
    float acc = 0.f;
    if (jv < Kk * Vp) {
        int j = jv / Vp, v = jv % Vp;
        for (int o = 0; o < En; ++o)
            acc += W0[((size_t)e * En + o) * Kk + j] * Wl[(size_t)v * En + o];
    }
    WcT[(size_t)jv * En + e] = (_Float16)acc;
}

__global__ void k_build_bc(const float* __restrict__ b0, const float* __restrict__ Wl,
                           const float* __restrict__ bl, float* __restrict__ bc) {
    int v = threadIdx.x;
    if (v >= Vp) return;
    float acc = bl[v];
    for (int o = 0; o < En; ++o) acc += b0[o] * Wl[(size_t)v * En + o];
    bc[v] = acc;
}

// ------------------------------------------- selection + bucketing (1 block/batch)
__global__ __launch_bounds__(NT) void k_select(const int* __restrict__ value,
                                               const int* __restrict__ depth,
                                               const int* __restrict__ maxd,
                                               int* __restrict__ sel,
                                               int* __restrict__ mlist,
                                               int* __restrict__ tlist,
                                               int4* __restrict__ tiles,
                                               int* __restrict__ tileCnt) {
    int b = blockIdx.x;
    int tid = threadIdx.x, lane = tid & 31, wid = tid >> 5;
    __shared__ int rbuf[NT];
    __shared__ int woff[NT / 32];
    __shared__ int sh_base;
    __shared__ int bcnt[8], bcur[8], boff[9];

    int target = *maxd - 1;
    // len_1[b] = count(depth[b,:] == max_depth-1)
    int c = 0;
    for (int i = tid; i < TTn; i += NT)
        c += (depth[(size_t)b * TTn + i] == target) ? 1 : 0;
    rbuf[tid] = c; __syncthreads();
    for (int s = NT / 2; s > 0; s >>= 1) {
        if (tid < s) rbuf[tid] += rbuf[tid + s];
        __syncthreads();
    }
    int len1 = rbuf[0];
    if (tid == 0) sh_base = 0;
    __syncthreads();

    // stable compaction of indices i < T1 with mask (i < len1 && value==2)
    for (int chunk = 0; chunk < T1n / NT; ++chunk) {
        int i = chunk * NT + tid;
        bool pred = (i < len1) && (value[(size_t)b * TTn + i] == 2);
        unsigned bal = (unsigned)__ballot((int)pred);
        int lpre = __popc(bal & ((1u << lane) - 1u));
        if (lane == 0) woff[wid] = __popc(bal);
        __syncthreads();
        if (tid == 0) {
            int s = sh_base;
            for (int w = 0; w < NT / 32; ++w) { int t = woff[w]; woff[w] = s; s += t; }
            sh_base = s;
        }
        __syncthreads();
        if (pred) {
            int p = woff[wid] + lpre;
            if (p < Mm) sel[b * Mm + p] = i;
        }
        __syncthreads();
    }
    int cnt = sh_base < Mm ? sh_base : Mm;   // rows >= cnt stay zero in x0h

    // bucket by j = sel % 8 (order within bucket is irrelevant: m travels along)
    if (tid < 8) bcnt[tid] = 0;
    __syncthreads();
    for (int m = tid; m < cnt; m += NT) atomicAdd(&bcnt[sel[b * Mm + m] & 7], 1);
    __syncthreads();
    if (tid == 0) {
        int s = 0;
        for (int j = 0; j < 8; ++j) { boff[j] = s; bcur[j] = 0; s += bcnt[j]; }
        boff[8] = s;
    }
    __syncthreads();
    for (int m = tid; m < cnt; m += NT) {
        int sv = sel[b * Mm + m];
        int j = sv & 7;
        int p = atomicAdd(&bcur[j], 1);
        mlist[b * Mm + boff[j] + p] = m;
        tlist[b * Mm + boff[j] + p] = sv >> 3;
    }
    __syncthreads();
    if (tid == 0) {
        int nt_total = 0;
        for (int j = 0; j < 8; ++j) nt_total += (bcnt[j] + 15) >> 4;
        int base = atomicAdd(tileCnt, nt_total);
        int idx = 0;
        for (int j = 0; j < 8; ++j) {
            int sz = bcnt[j];
            int ntl = (sz + 15) >> 4;
            for (int t = 0; t < ntl; ++t) {
                int4 d;
                d.x = b; d.y = j;
                d.z = b * Mm + boff[j] + t * 16;   // abs start in mlist/tlist
                d.w = b * Mm + boff[j] + sz;       // abs end
                tiles[base + idx++] = d;
            }
        }
    }
}

// ---------------------------------------------------------------- stage 1 GEMM
// x_0[b][m][:] = x[b][t][:] @ W1[:,:,j] + b1  for each bucket tile (16 rows)
__global__ __launch_bounds__(128) void k_stage1(const _Float16* __restrict__ xh,
                                                const _Float16* __restrict__ Whb,
                                                const float* __restrict__ b1,
                                                const int4* __restrict__ tiles,
                                                const int* __restrict__ tileCnt,
                                                const int* __restrict__ mlist,
                                                const int* __restrict__ tlist,
                                                _Float16* __restrict__ x0h) {
    if ((int)blockIdx.x >= *tileCnt) return;
    int4 d = tiles[blockIdx.x];
    int b = d.x, j = d.y, s = d.z, e = d.w;

    __shared__ _Float16 Ash[16 * En];
    __shared__ int sh_m[16], sh_t[16];
    int tid = threadIdx.x;
    if (tid < 16) {
        int r = s + tid, m = -1, t = 0;
        if (r < e) { m = mlist[r]; t = tlist[r]; }
        sh_m[tid] = m; sh_t[tid] = t;
    }
    __syncthreads();
    {   // stage gathered A tile (16 rows x 256 f16) into LDS
        int row = tid >> 3;
        int h0 = (tid & 7) * 32;
        const uint4* src = (const uint4*)(xh + ((size_t)b * Tcn + sh_t[row]) * En + h0);
        uint4* dst = (uint4*)(Ash + row * En + h0);
        dst[0] = src[0]; dst[1] = src[1]; dst[2] = src[2]; dst[3] = src[3];
    }
    __syncthreads();

    int wave = tid >> 5, lane = tid & 31;
    int mrow = lane & 15, hi = lane >> 4;
    const _Float16* Wb = Whb + (size_t)j * En * En;

    v8f acc[4] = {};
#pragma unroll
    for (int ks = 0; ks < En / 32; ++ks) {
        int K0 = ks * 32;
        union { v16h v; uint4 q[2]; } ua;       // A 16x32 f16 fragment layout
        const _Float16* ar = Ash + mrow * En + K0 + 16 * hi;
        ua.q[0] = *(const uint4*)ar;            // K = K0+16*hi .. +7
        ua.q[1] = *(const uint4*)(ar + 8);      // K = K0+16*hi+8 .. +7
#pragma unroll
        for (int q = 0; q < 4; ++q) {
            int N0 = (wave * 4 + q) * 16;
            v16h bv = *(const v16h*)(Wb + (size_t)(N0 + mrow) * En + K0 + 16 * hi);
            acc[q] = __builtin_amdgcn_wmma_f32_16x16x32_f16(
                false, ua.v, false, bv, (short)0, acc[q], false, false);
        }
    }
#pragma unroll
    for (int q = 0; q < 4; ++q) {
        int N0 = (wave * 4 + q) * 16;
        float bb = b1[N0 + mrow];
#pragma unroll
        for (int v = 0; v < 8; ++v) {
            int row = v + 8 * hi;
            int m = sh_m[row];
            if (m >= 0)
                x0h[((size_t)b * Mm + m) * En + N0 + mrow] = (_Float16)(acc[q][v] + bb);
        }
    }
}

// --------------------------------------------- stage 2 GEMM (fused deconv0+head)
// out[b][m*8+j][v] = x_0[b][m][:] . WcT[j*17+v][:] + bc[v]
__global__ __launch_bounds__(32) void k_stage2(const _Float16* __restrict__ x0h,
                                               const _Float16* __restrict__ WcT,
                                               const float* __restrict__ bc,
                                               float* __restrict__ out) {
    int gb = blockIdx.x;
    int b = gb / (Mm / 16);
    int m0 = (gb % (Mm / 16)) * 16;
    int lane = threadIdx.x & 31;
    int mrow = lane & 15, hi = lane >> 4;

    v8f acc[9] = {};
    const _Float16* arowBase = x0h + ((size_t)b * Mm + m0 + mrow) * En;
#pragma unroll
    for (int ks = 0; ks < En / 32; ++ks) {
        int K0 = ks * 32;
        union { v16h v; uint4 q[2]; } ua;
        const _Float16* ar = arowBase + K0 + 16 * hi;
        ua.q[0] = *(const uint4*)ar;
        ua.q[1] = *(const uint4*)(ar + 8);
#pragma unroll
        for (int nt = 0; nt < 9; ++nt) {
            v16h bv = *(const v16h*)(WcT + (size_t)(nt * 16 + mrow) * En + K0 + 16 * hi);
            acc[nt] = __builtin_amdgcn_wmma_f32_16x16x32_f16(
                false, ua.v, false, bv, (short)0, acc[nt], false, false);
        }
    }
#pragma unroll
    for (int nt = 0; nt < 9; ++nt) {
        int jv = nt * 16 + mrow;
        if (jv < Kk * Vp) {
            int j = jv / Vp, vs = jv % Vp;
            float bcv = bc[vs];
#pragma unroll
            for (int v = 0; v < 8; ++v) {
                int m = m0 + v + 8 * hi;
                out[(((size_t)b * Mm + m) * Kk + j) * Vp + vs] = acc[nt][v] + bcv;
            }
        }
    }
}

// --------------------------------------------------------------------- launch
extern "C" void kernel_launch(void* const* d_in, const int* in_sizes, int n_in,
                              void* d_out, int out_size, void* d_ws, size_t ws_size,
                              hipStream_t stream) {
    const float* x     = (const float*)d_in[0];
    const int*   value = (const int*)d_in[1];
    const int*   depth = (const int*)d_in[2];
    // d_in[3] = pos (unused by reference), d_in[4] = num_mix (== Mm, compile-time)
    const float* W1 = (const float*)d_in[5];
    const float* b1 = (const float*)d_in[6];
    const float* W0 = (const float*)d_in[7];
    const float* b0 = (const float*)d_in[8];
    const float* Wl = (const float*)d_in[9];
    const float* bl = (const float*)d_in[10];
    float* out = (float*)d_out;

    char* ws = (char*)d_ws;
    size_t off = 0;
    auto alloc = [&](size_t bytes) -> char* {
        char* p = ws + off;
        off = (off + bytes + 255) & ~(size_t)255;
        return p;
    };
    int*       tileCnt = (int*)alloc(8);
    int*       maxd    = (int*)alloc(8);
    _Float16*  xh      = (_Float16*)alloc((size_t)Bn * Tcn * En * 2);
    _Float16*  Whb     = (_Float16*)alloc((size_t)Kk * En * En * 2);
    _Float16*  WcT     = (_Float16*)alloc((size_t)NPAD * En * 2);
    float*     bc      = (float*)alloc(Vp * 4);
    _Float16*  x0h     = (_Float16*)alloc((size_t)Bn * Mm * En * 2);
    int*       sel     = (int*)alloc((size_t)Bn * Mm * 4);
    int*       mlist   = (int*)alloc((size_t)Bn * Mm * 4);
    int*       tlist   = (int*)alloc((size_t)Bn * Mm * 4);
    int4*      tiles   = (int4*)alloc((size_t)MAXTILES * 16);

    k_init<<<512, 256, 0, stream>>>((uint32_t*)x0h, tileCnt);
    k_maxdepth<<<1, 1024, 0, stream>>>(depth, maxd);
    k_cvt_x<<<2048, 256, 0, stream>>>(x, xh);
    k_cvt_w1<<<1024, 256, 0, stream>>>(W1, Whb);
    k_build_wc<<<(NPAD * En + 255) / 256, 256, 0, stream>>>(W0, Wl, WcT);
    k_build_bc<<<1, 32, 0, stream>>>(b0, Wl, bl, bc);
    k_select<<<Bn, NT, 0, stream>>>(value, depth, maxd, sel, mlist, tlist, tiles, tileCnt);
    k_stage1<<<MAXTILES, 128, 0, stream>>>(xh, Whb, b1, tiles, tileCnt, mlist, tlist, x0h);
    k_stage2<<<Bn * Mm / 16, 32, 0, stream>>>(x0h, WcT, bc, out);
}
